// Transformer_Module_68504728371238
// MI455X (gfx1250) — matrix operand order repaired and checked by reading, not compiled, not run on hardware
//
#include <hip/hip_runtime.h>
#include <hip/hip_bf16.h>

// MI455X / gfx1250 implementation of the 4-layer prefix-mask transformer stack.
// 1024 blocks = (128 masks) x (8 batch). Each block runs the whole 4-layer
// transformer for its sequence entirely in LDS (h stays on-chip, ~319KB LDS),
// streaming bf16-packed weights from L2 (weights = 10MB bf16, fully L2-resident).
// Matrix math: v_wmma_f32_16x16x32_bf16 throughout.

typedef __bf16 bf16;
typedef __attribute__((ext_vector_type(16))) __bf16 v16bf;
typedef __attribute__((ext_vector_type(8)))  __bf16 v8bf;
typedef __attribute__((ext_vector_type(8)))  float  v8f;

#define DEVI static __device__ __forceinline__

// Problem constants
constexpr int E  = 512;
constexpr int H  = 8;
constexpr int HD = 64;
constexpr int S  = 128;
constexpr int NL = 4;
constexpr int KT = 16;   // 512 / 32 k-steps per GEMM column tile
constexpr int NT = 32;   // 512 / 16 column tiles

// LDS layout (bf16 element offsets). Strides chosen 16B-aligned + bank-skewed.
constexpr int SH = 520;                    // h / o row stride
constexpr int SQ = 64;                     // Q row stride
constexpr int SK = 72;                     // K row stride
constexpr int SV = 136;                    // V^T row stride (rows = head dim)
constexpr int SP = 32;                     // per-wave P chunk row stride
constexpr int OFF_H = 0;
constexpr int OFF_O = OFF_H + S * SH;      // 66560
constexpr int OFF_Q = OFF_O + S * SH;      // 133120
constexpr int OFF_K = OFF_Q + S * SQ;      // 141312
constexpr int OFF_V = OFF_K + S * SK;      // 150528
constexpr int OFF_P = OFF_V + HD * SV;     // 159232
constexpr int SMEM_ELEMS = OFF_P + 8 * 16 * SP;   // 163328 elems = 326656 B

DEVI v8f wmma_bf16(v16bf a, v16bf b, v8f c) {
  // D(16x16 f32) += A(16x32 bf16) * B(32x16 bf16)
  return __builtin_amdgcn_wmma_f32_16x16x32_bf16(false, a, false, b,
                                                 (short)0, c, false, false);
}

// A fragment from row-major LDS buffer [row][k]:
// lane L: row = row0 + L%16 ; k = k0 + (L/16)*16 + j (16 consecutive)
DEVI v16bf load_a(const bf16* buf, int stride, int row0, int k0, int lane) {
  const int g = (lane >> 4) & 1, m = lane & 15;
  const bf16* p = buf + (row0 + m) * stride + k0 + g * 16;
  v8bf lo = *(const v8bf*)(p);
  v8bf hi = *(const v8bf*)(p + 8);
  v16bf a;
#pragma unroll
  for (int j = 0; j < 8; ++j) { a[j] = lo[j]; a[j + 8] = hi[j]; }
  return a;
}

// B fragment from row-major LDS buffer [n][k]:
// lane L: n = n0 + L%16 ; k = k0 + (L/16)*16 + j (16 consecutive)
DEVI v16bf load_b(const bf16* buf, int stride, int n0, int k0, int lane) {
  const bf16* p = buf + (n0 + (lane & 15)) * stride + k0 + ((lane >> 4) & 1) * 16;
  v8bf lo = *(const v8bf*)(p);
  v8bf hi = *(const v8bf*)(p + 8);
  v16bf b;
#pragma unroll
  for (int j = 0; j < 8; ++j) { b[j] = lo[j]; b[j + 8] = hi[j]; }
  return b;
}

// B fragment from pre-packed global weights: one aligned 32B load per lane.
DEVI v16bf load_bw(const bf16* __restrict__ w, int tile, int lane) {
  return *(const v16bf*)(w + (size_t)tile * 512 + lane * 16);
}

// C tile (v8f) -> row-major LDS bf16. element r: row = row0 + (L/16)*8 + r, col = col0 + L%16
DEVI void store_c(bf16* buf, int stride, int row0, int col0, v8f c, int lane) {
  const int g = (lane >> 4) & 1, m = lane & 15;
#pragma unroll
  for (int r = 0; r < 8; ++r)
    buf[(row0 + g * 8 + r) * stride + col0 + m] = (bf16)c[r];
}

// C tile stored transposed: value(row,col) -> buf[col][row]   (used for V^T)
DEVI void store_cT(bf16* buf, int stride, int row0, int col0, v8f c, int lane) {
  const int g = (lane >> 4) & 1, m = lane & 15;
#pragma unroll
  for (int r = 0; r < 8; ++r)
    buf[(col0 + m) * stride + row0 + g * 8 + r] = (bf16)c[r];
}

// One 16x16 output tile of X(rows x 512) @ Wpacked(512 x 512), column tile nt.
DEVI v8f gemm512(const bf16* X, int xs, int row0,
                 const bf16* __restrict__ w, int nt, int lane) {
  v8f acc = {0.f, 0.f, 0.f, 0.f, 0.f, 0.f, 0.f, 0.f};
#pragma unroll
  for (int kt = 0; kt < KT; ++kt) {
    v16bf a = load_a(X, xs, row0, kt * 32, lane);
    v16bf b = load_bw(w, nt * KT + kt, lane);
    acc = wmma_bf16(a, b, acc);
  }
  return acc;
}

// Row-wise LayerNorm over 16 wave-private rows (512 cols); src may equal dst.
DEVI void layernorm_rows(const bf16* src, bf16* dst, int row0, int lane,
                         const float* __restrict__ gw, const float* __restrict__ bw) {
  float gv[16], bv[16];
#pragma unroll
  for (int t = 0; t < 16; ++t) { gv[t] = gw[lane + 32 * t]; bv[t] = bw[lane + 32 * t]; }
#pragma unroll 1
  for (int r = 0; r < 16; ++r) {
    const bf16* pr = src + (row0 + r) * SH;
    float vals[16], s = 0.f, ss = 0.f;
#pragma unroll
    for (int t = 0; t < 16; ++t) {
      float v = (float)pr[lane + 32 * t];
      vals[t] = v; s += v; ss += v * v;
    }
#pragma unroll
    for (int msk = 1; msk < 32; msk <<= 1) {
      s  += __shfl_xor(s,  msk);
      ss += __shfl_xor(ss, msk);
    }
    float mean = s * (1.f / 512.f);
    float var  = ss * (1.f / 512.f) - mean * mean;
    float rstd = rsqrtf(var + 1e-5f);
    bf16* pw = dst + (row0 + r) * SH;
#pragma unroll
    for (int t = 0; t < 16; ++t)
      pw[lane + 32 * t] = (bf16)((vals[t] - mean) * rstd * gv[t] + bv[t]);
  }
}

// ---------------------------------------------------------------------------
// Pre-pass: fp32 weights -> bf16 in WMMA B-fragment order
// layout: [L][5][nt(32)][kt(16)][lane(32)][j(16)]
//   n = nt*16 + lane%16 ;  k = kt*32 + (lane/16)*16 + j
// ---------------------------------------------------------------------------
__global__ __launch_bounds__(256)
void pack_weights(const float* __restrict__ Wq, const float* __restrict__ Wk,
                  const float* __restrict__ Wv, const float* __restrict__ Wo,
                  const float* __restrict__ Wf, bf16* __restrict__ out) {
  const size_t TOT = (size_t)NL * 5 * E * E;
  size_t idx = (size_t)blockIdx.x * 256 + threadIdx.x;
  if (idx >= TOT) return;
  int j  = (int)(idx & 15);
  size_t t = idx >> 4;
  int ln = (int)(t & 31); t >>= 5;
  int kt = (int)(t & 15); t >>= 4;
  int nt = (int)(t & 31); t >>= 5;
  int mat = (int)(t % 5);
  int l   = (int)(t / 5);
  int n = nt * 16 + (ln & 15);
  int k = kt * 32 + (ln >> 4) * 16 + j;
  const float* src = (mat == 0) ? Wq : (mat == 1) ? Wk : (mat == 2) ? Wv
                   : (mat == 3) ? Wo : Wf;
  out[idx] = (bf16)src[((size_t)l * E + k) * E + n];
}

// ---------------------------------------------------------------------------
// Main kernel: one block = one (mask, batch) sequence, 8 wave32 (256 thr).
// Wave w owns rows [16w, 16w+16): Q, residual, LN, O-proj, FFN are wave-private;
// only K/V per head are shared -> exactly 2 barriers per head.
// ---------------------------------------------------------------------------
__global__ __launch_bounds__(256)
void txf_stack(const float* __restrict__ x,  const bf16* __restrict__ Wp,
               const float* __restrict__ bq, const float* __restrict__ bk,
               const float* __restrict__ bv, const float* __restrict__ bo,
               const float* __restrict__ bff,
               const float* __restrict__ g1, const float* __restrict__ b1,
               const float* __restrict__ g2, const float* __restrict__ b2,
               float* __restrict__ out) {
  extern __shared__ bf16 smem[];
  bf16* sh_h = smem + OFF_H;
  bf16* sh_o = smem + OFF_O;
  bf16* sh_q = smem + OFF_Q;
  bf16* sh_k = smem + OFF_K;
  bf16* sh_v = smem + OFF_V;

  const int maskI = blockIdx.x;          // prefix mask index 0..127
  const int batch = blockIdx.y;          // 0..7
  const int npos  = maskI + 1;           // valid key count
  const int tid  = threadIdx.x;
  const int wave = tid >> 5;
  const int lane = tid & 31;
  const int g    = (lane >> 4) & 1;
  const int m16  = lane & 15;
  const int row0 = wave * 16;
  bf16* sh_pw = smem + OFF_P + wave * 16 * SP;

  // Load x rows (fp32 -> bf16) into wave-private row block of sh_h.
  const float* xb = x + (size_t)batch * S * E;
#pragma unroll 1
  for (int r = 0; r < 16; ++r) {
    const float* xr = xb + (size_t)(row0 + r) * E;
    bf16* hr = sh_h + (row0 + r) * SH;
    for (int e = lane; e < E; e += 32) hr[e] = (bf16)xr[e];
  }

#pragma unroll 1
  for (int l = 0; l < NL; ++l) {
    const bf16* Wq = Wp + (size_t)(l * 5 + 0) * (E * E);
    const bf16* Wk = Wp + (size_t)(l * 5 + 1) * (E * E);
    const bf16* Wv = Wp + (size_t)(l * 5 + 2) * (E * E);
    const bf16* Wo = Wp + (size_t)(l * 5 + 3) * (E * E);
    const bf16* Wf = Wp + (size_t)(l * 5 + 4) * (E * E);

    // ================= attention, head by head =================
#pragma unroll 1
    for (int hh = 0; hh < H; ++hh) {
      // Q/K/V for this wave's 16 rows, 4 column tiles of this head
#pragma unroll 1
      for (int ntl = 0; ntl < 4; ++ntl) {
        const int ntg = hh * 4 + ntl;
        const int col = ntg * 16 + m16;
        v8f cq = gemm512(sh_h, SH, row0, Wq, ntg, lane);
        v8f ck = gemm512(sh_h, SH, row0, Wk, ntg, lane);
        v8f cv = gemm512(sh_h, SH, row0, Wv, ntg, lane);
        const float aq = bq[l * E + col], ak = bk[l * E + col], av = bv[l * E + col];
#pragma unroll
        for (int r = 0; r < 8; ++r) { cq[r] += aq; ck[r] += ak; cv[r] += av; }
        store_c (sh_q, SQ, row0, ntl * 16, cq, lane);
        store_c (sh_k, SK, row0, ntl * 16, ck, lane);
        store_cT(sh_v, SV, row0, ntl * 16, cv, lane);
      }
      __syncthreads();   // K/V visible to all waves

      // ---- scores: 16 queries (own rows) x 128 keys, in registers
      v8f sc[8];
      {
        v16bf qa0 = load_a(sh_q, SQ, row0, 0,  lane);
        v16bf qa1 = load_a(sh_q, SQ, row0, 32, lane);
#pragma unroll
        for (int kt = 0; kt < 8; ++kt) {
          v8f a = {0.f, 0.f, 0.f, 0.f, 0.f, 0.f, 0.f, 0.f};
          a = wmma_bf16(qa0, load_b(sh_k, SK, kt * 16, 0,  lane), a);
          a = wmma_bf16(qa1, load_b(sh_k, SK, kt * 16, 32, lane), a);
          const int key = kt * 16 + m16;
#pragma unroll
          for (int r = 0; r < 8; ++r)
            a[r] = (key < npos) ? a[r] * 0.125f : -1.0e9f;   // 1/sqrt(64), prefix mask
          sc[kt] = a;
        }
      }
      // ---- softmax (row = g*8 + r, spread over 16 lanes x 8 tiles)
      float rsum[8];
#pragma unroll
      for (int r = 0; r < 8; ++r) {
        float mx = sc[0][r];
#pragma unroll
        for (int kt = 1; kt < 8; ++kt) mx = fmaxf(mx, sc[kt][r]);
        mx = fmaxf(mx, __shfl_xor(mx, 1));
        mx = fmaxf(mx, __shfl_xor(mx, 2));
        mx = fmaxf(mx, __shfl_xor(mx, 4));
        mx = fmaxf(mx, __shfl_xor(mx, 8));
        float s = 0.f;
#pragma unroll
        for (int kt = 0; kt < 8; ++kt) {
          float e = __expf(sc[kt][r] - mx);
          sc[kt][r] = e; s += e;
        }
        s += __shfl_xor(s, 1);
        s += __shfl_xor(s, 2);
        s += __shfl_xor(s, 4);
        s += __shfl_xor(s, 8);
        rsum[r] = 1.0f / s;
      }
      // ---- O = P @ V  (stage P through per-wave LDS chunk for A-layout)
      v8f oacc[4];
#pragma unroll
      for (int nt = 0; nt < 4; ++nt) {
        const v8f z = {0.f, 0.f, 0.f, 0.f, 0.f, 0.f, 0.f, 0.f};
        oacc[nt] = z;
      }
#pragma unroll 1
      for (int c = 0; c < 4; ++c) {
#pragma unroll
        for (int half = 0; half < 2; ++half) {
          const int kt = c * 2 + half;
#pragma unroll
          for (int r = 0; r < 8; ++r)
            sh_pw[(g * 8 + r) * SP + half * 16 + m16] = (bf16)sc[kt][r];
        }
        v16bf pa = load_a(sh_pw, SP, 0, 0, lane);   // same-wave LDS RAW: HW waits
#pragma unroll
        for (int nt = 0; nt < 4; ++nt) {
          v16bf vb = load_b(sh_v, SV, nt * 16, c * 32, lane);
          oacc[nt] = wmma_bf16(pa, vb, oacc[nt]);
        }
      }
      // normalize + write this head's 64 columns of o (own rows)
#pragma unroll
      for (int nt = 0; nt < 4; ++nt)
#pragma unroll
        for (int r = 0; r < 8; ++r)
          sh_o[(row0 + g * 8 + r) * SH + hh * HD + nt * 16 + m16] =
              (bf16)(oacc[nt][r] * rsum[r]);
      __syncthreads();   // all waves done with this head's K/V
    }

    // ================= o @ Wo + bo + h -> LN1 (wave-private rows) ==========
#pragma unroll 1
    for (int nt = 0; nt < NT; ++nt) {
      v8f c = gemm512(sh_o, SH, row0, Wo, nt, lane);
      const float bb = bo[l * E + nt * 16 + m16];
#pragma unroll
      for (int r = 0; r < 8; ++r) {
        bf16* hp = sh_h + (row0 + g * 8 + r) * SH + nt * 16 + m16;
        *hp = (bf16)(c[r] + bb + (float)(*hp));   // residual, read-then-write
      }
    }
    layernorm_rows(sh_h, sh_h, row0, lane, g1 + l * E, b1 + l * E);

    // ================= relu(h @ Wf + bf) + h -> LN2 ========================
#pragma unroll 1
    for (int nt = 0; nt < NT; ++nt) {
      v8f c = gemm512(sh_h, SH, row0, Wf, nt, lane);
      const float bb = bff[l * E + nt * 16 + m16];
#pragma unroll
      for (int r = 0; r < 8; ++r) {
        const int off = (row0 + g * 8 + r) * SH + nt * 16 + m16;
        float t = fmaxf(c[r] + bb, 0.0f) + (float)sh_h[off];
        sh_o[off] = (bf16)t;
      }
    }
    layernorm_rows(sh_o, sh_h, row0, lane, g2 + l * E, b2 + l * E);
  }

  // Only row maskI survives: out[batch][maskI][:]  (row owned by wave maskI>>4)
  if (wave == (maskI >> 4)) {
    float* op = out + ((size_t)batch * S + maskI) * E;
    const bf16* hr = sh_h + maskI * SH;
    for (int e = lane; e < E; e += 32) op[e] = (float)hr[e];
  }
}

// ---------------------------------------------------------------------------
extern "C" void kernel_launch(void* const* d_in, const int* in_sizes, int n_in,
                              void* d_out, int out_size, void* d_ws, size_t ws_size,
                              hipStream_t stream) {
  const float* x  = (const float*)d_in[0];
  const float* Wq = (const float*)d_in[1];
  const float* bq = (const float*)d_in[2];
  const float* Wk = (const float*)d_in[3];
  const float* bk = (const float*)d_in[4];
  const float* Wv = (const float*)d_in[5];
  const float* bv = (const float*)d_in[6];
  const float* Wo = (const float*)d_in[7];
  const float* bo = (const float*)d_in[8];
  const float* Wf = (const float*)d_in[9];
  const float* bf_ = (const float*)d_in[10];
  const float* g1 = (const float*)d_in[11];
  const float* b1 = (const float*)d_in[12];
  const float* g2 = (const float*)d_in[13];
  const float* b2 = (const float*)d_in[14];

  bf16* Wp = (bf16*)d_ws;   // 10.5 MB packed bf16 weights
  const size_t TOT = (size_t)NL * 5 * E * E;
  pack_weights<<<(unsigned)((TOT + 255) / 256), 256, 0, stream>>>(Wq, Wk, Wv, Wo, Wf, Wp);

  dim3 grid(S, 8);          // 128 masks x 8 batch
  const size_t lds_bytes = (size_t)SMEM_ELEMS * sizeof(bf16);   // 326,656 B
  txf_stack<<<grid, 256, lds_bytes, stream>>>(x, Wp, bq, bk, bv, bo, bf_,
                                              g1, b1, g2, b2, (float*)d_out);
}
